// RelativeMultiHeadAttn_38156489457867
// MI455X (gfx1250) — compile-verified, matrix operand order checked by reading, not run on hardware
//
#include <hip/hip_runtime.h>
#include <hip/hip_bf16.h>
#include <math.h>

// Problem constants (match reference)
constexpr int NB = 2;      // batch
constexpr int L  = 2048;   // sequence length
constexpr int DM = 512;    // model dim
constexpr int NH = 8;      // heads
constexpr int HD = 64;     // head dim
constexpr int TS = 168;    // padded LDS stride for rel slab (160 cols used)

typedef __attribute__((ext_vector_type(16))) __bf16 v16bf;
typedef __attribute__((ext_vector_type(8)))  float  v8f;

// ---------------------------------------------------------------------------
// WMMA helpers (CDNA5 wave32, 16x16x32 bf16 -> f32)
// ---------------------------------------------------------------------------
__device__ __forceinline__ v8f wmma_bf16(v16bf a, v16bf b, v8f c) {
    // 8 args: (neg_a, A, neg_b, B, c_mod, C, reuse_a, reuse_b)
    return __builtin_amdgcn_wmma_f32_16x16x32_bf16(false, a, false, b,
                                                   (short)0, c, false, false);
}

// A fragment: lane holds row M = lane%16, 16 consecutive K at (lane/16)*16.
// B fragment is identical when the operand is stored [N, K] row-major.
__device__ __forceinline__ v16bf load_frag_bf16(const __bf16* base, int stride,
                                                int row0, int k0) {
    int lane = threadIdx.x & 31;
    const __bf16* p = base + (size_t)(row0 + (lane & 15)) * stride
                           + k0 + ((lane >> 4) << 4);
    return *(const v16bf*)p;                 // contiguous 32B per lane
}

// Same, from fp32 source with convert (16 floats = 64B contiguous per lane).
__device__ __forceinline__ v16bf load_frag_f32(const float* base, int stride,
                                               int row0, int k0) {
    int lane = threadIdx.x & 31;
    const float4* p = (const float4*)(base + (size_t)(row0 + (lane & 15)) * stride
                                           + k0 + ((lane >> 4) << 4));
    v16bf o;
#pragma unroll
    for (int t = 0; t < 4; ++t) {
        float4 f = p[t];
        o[4*t+0] = (__bf16)f.x; o[4*t+1] = (__bf16)f.y;
        o[4*t+2] = (__bf16)f.z; o[4*t+3] = (__bf16)f.w;
    }
    return o;
}

// r-row fragment with range clamp (rows >= 2L read as zero; never actually used)
__device__ __forceinline__ v16bf load_r_frag(const __bf16* rbf, int n0, int k0) {
    int lane = threadIdx.x & 31;
    int row  = n0 + (lane & 15);
    bool ok  = (unsigned)row < (unsigned)(2 * L);
    const __bf16* p = rbf + (size_t)(ok ? row : 0) * HD + k0 + ((lane >> 4) << 4);
    v16bf v = *(const v16bf*)p;
    if (!ok) {
#pragma unroll
        for (int t = 0; t < 16; ++t) v[t] = (__bf16)0.0f;
    }
    return v;
}

// ---------------------------------------------------------------------------
// Kernel 1: sinusoid table row m, then r[m,:] = pos[m,:] @ Wr^T  (bf16 out)
// ---------------------------------------------------------------------------
__global__ __launch_bounds__(64)
void build_r_kernel(const float* __restrict__ Wr, __bf16* __restrict__ rbf) {
    __shared__ float pos[HD];
    int m = blockIdx.x;          // 0 .. 2L-1  (positions = table rows directly)
    int d = threadIdx.x;         // 0 .. 63
    const float inv = 0.29710775393471563f;   // log(10000)/31
    float val = 0.0f;
    if (m != 0) {                // padding_idx row is zeroed
        int j = d & 31;
        float ang = (float)m * __expf(-(float)j * inv);
        val = (d < 32) ? __sinf(ang) : __cosf(ang);
    }
    pos[d] = val;
    __syncthreads();
    float acc = 0.0f;
#pragma unroll
    for (int kk = 0; kk < HD; ++kk) acc += pos[kk] * Wr[d * HD + kk];
    rbf[(size_t)m * HD + d] = (__bf16)acc;
}

// ---------------------------------------------------------------------------
// Kernel 2: projections. qp = q@Wq^T -> q1/q2 (+rel biases); kv = k@Wkv^T ->
// k (head-major) and v stored TRANSPOSED [b,h,64,L] for contiguous B-fragments.
// 96 16-wide column tiles per (b, 16-row tile); one tile per wave, WMMA K=512.
// ---------------------------------------------------------------------------
__global__ __launch_bounds__(256)
void proj_kernel(const float* __restrict__ q, const float* __restrict__ k,
                 const float* __restrict__ Wq, const float* __restrict__ Wkv,
                 const float* __restrict__ rrb, const float* __restrict__ rwb,
                 __bf16* __restrict__ q1, __bf16* __restrict__ q2,
                 __bf16* __restrict__ kb, __bf16* __restrict__ vt) {
    int b    = blockIdx.z;
    int i0   = blockIdx.y * 16;
    int wave = threadIdx.x >> 5;
    int lane = threadIdx.x & 31;
    int t    = blockIdx.x * 8 + wave;          // 0..95 (wave-uniform)
    int N = lane & 15, hf = lane >> 4;

    v8f c = {};
    if (t < 32) {                               // ---- qp tile
        int n0 = t * 16;
        const float* A = q + (size_t)b * L * DM;
        for (int k0 = 0; k0 < DM; k0 += 32) {
            v16bf a = load_frag_f32(A,  DM, i0, k0);
            v16bf w = load_frag_f32(Wq, DM, n0, k0);   // Wq row n = (Wq^T) col n
            c = wmma_bf16(a, w, c);
        }
        int col = n0 + N, h = col >> 6, d = col & 63;
        float b1 = rrb[h * HD + d], b2 = rwb[h * HD + d];
        size_t base = ((size_t)(b * NH + h) * L) * HD + d;
#pragma unroll
        for (int e = 0; e < 8; ++e) {
            int i = i0 + e + (hf << 3);
            q1[base + (size_t)i * HD] = (__bf16)(c[e] + b1);
            q2[base + (size_t)i * HD] = (__bf16)(c[e] + b2);
        }
    } else {                                    // ---- kv tile
        int n0 = (t - 32) * 16;
        const float* A = k + (size_t)b * L * DM;
        for (int k0 = 0; k0 < DM; k0 += 32) {
            v16bf a = load_frag_f32(A,   DM, i0, k0);
            v16bf w = load_frag_f32(Wkv, DM, n0, k0);
            c = wmma_bf16(a, w, c);
        }
        int col = n0 + N;
        if (col < DM) {                         // K half (uniform per tile)
            int h = col >> 6, d = col & 63;
            size_t base = ((size_t)(b * NH + h) * L) * HD + d;
#pragma unroll
            for (int e = 0; e < 8; ++e) {
                int i = i0 + e + (hf << 3);
                kb[base + (size_t)i * HD] = (__bf16)c[e];
            }
        } else {                                // V half, transposed [h][d][i]
            int cc = col - DM, h = cc >> 6, d = cc & 63;
            size_t base = ((size_t)(b * NH + h) * HD + d) * L;
#pragma unroll
            for (int e = 0; e < 8; ++e) {
                int i = i0 + e + (hf << 3);
                vt[base + i] = (__bf16)c[e];
            }
        }
    }
}

// ---------------------------------------------------------------------------
// Kernel 3: raw scores. Block = (b,h, 16-row i-tile, 128-col j-chunk), 8 waves.
// Phase 1: T[16 x 160] = q2_tile @ r^T for m in [mbase, mbase+160) -> LDS.
// Phase 2: each wave does one 16x16 QK^T WMMA tile, adds skew-gathered rel
// term T[M][(L+j-i)-mbase], scales by 1/8, writes raw scores to attn.
// ---------------------------------------------------------------------------
__global__ __launch_bounds__(256)
void scores_kernel(const __bf16* __restrict__ q1, const __bf16* __restrict__ q2,
                   const __bf16* __restrict__ kb, const __bf16* __restrict__ rbf,
                   float* __restrict__ attn) {
    __shared__ float Tlds[16 * TS];
    int bh   = blockIdx.z;
    int i0   = blockIdx.y * 16;
    int j0   = blockIdx.x * 128;
    int wave = threadIdx.x >> 5;
    int lane = threadIdx.x & 31;
    int N = lane & 15, hf = lane >> 4;

    const __bf16* q1p = q1 + (size_t)bh * L * HD;
    const __bf16* q2p = q2 + (size_t)bh * L * HD;
    const __bf16* kp  = kb + (size_t)bh * L * HD;

    v16bf a20 = load_frag_bf16(q2p, HD, i0, 0);
    v16bf a21 = load_frag_bf16(q2p, HD, i0, 32);

    int mbase = L + j0 - i0 - 15;               // needed m range: 160 wide
    for (int t = wave; t < 10; t += 8) {        // 10 column tiles over 8 waves
        int n0 = mbase + t * 16;
        v16bf b0 = load_r_frag(rbf, n0, 0);
        v16bf b1 = load_r_frag(rbf, n0, 32);
        v8f c = {};
        c = wmma_bf16(a20, b0, c);
        c = wmma_bf16(a21, b1, c);
#pragma unroll
        for (int e = 0; e < 8; ++e)
            Tlds[(e + (hf << 3)) * TS + t * 16 + N] = c[e];
    }
    __syncthreads();

    int jw = j0 + wave * 16;
    v16bf a10 = load_frag_bf16(q1p, HD, i0, 0);
    v16bf a11 = load_frag_bf16(q1p, HD, i0, 32);
    v16bf k0f = load_frag_bf16(kp, HD, jw, 0);  // k rows = (k^T) columns
    v16bf k1f = load_frag_bf16(kp, HD, jw, 32);
    v8f c = {};
    c = wmma_bf16(a10, k0f, c);
    c = wmma_bf16(a11, k1f, c);

#pragma unroll
    for (int e = 0; e < 8; ++e) {
        int M = e + (hf << 3);
        int i = i0 + M, j = jw + N;
        int idx = wave * 16 + N - M + 15;       // == (L + j - i) - mbase
        float s = (c[e] + Tlds[M * TS + idx]) * 0.125f;   // 1/sqrt(64)
        attn[((size_t)bh * L + i) * L + j] = s;
    }
}

// ---------------------------------------------------------------------------
// Kernel 4: masked row softmax over L=2048, in place (one block per row).
// ---------------------------------------------------------------------------
__global__ __launch_bounds__(256)
void softmax_kernel(float* __restrict__ attn, const int* __restrict__ mask) {
    __shared__ float red[256];
    int row = blockIdx.x;                 // (b*NH + h)*L + i
    int i = row % L;
    int b = row / (NH * L);
    float* p = attn + (size_t)row * L;
    const int* mrow = mask + ((size_t)b * L + i) * L;
    int tid = threadIdx.x;

    float v[8];
    float mx = -INFINITY;
#pragma unroll
    for (int t = 0; t < 8; ++t) {
        int j = tid + t * 256;
        float s = p[j];
        if (mrow[j] == 0) s = -INFINITY;
        v[t] = s;
        mx = fmaxf(mx, s);
    }
    red[tid] = mx; __syncthreads();
    for (int o = 128; o > 0; o >>= 1) {
        if (tid < o) red[tid] = fmaxf(red[tid], red[tid + o]);
        __syncthreads();
    }
    mx = red[0]; __syncthreads();

    float sum = 0.0f;
#pragma unroll
    for (int t = 0; t < 8; ++t) { float e = __expf(v[t] - mx); v[t] = e; sum += e; }
    red[tid] = sum; __syncthreads();
    for (int o = 128; o > 0; o >>= 1) {
        if (tid < o) red[tid] += red[tid + o];
        __syncthreads();
    }
    float inv = 1.0f / red[0];
#pragma unroll
    for (int t = 0; t < 8; ++t) p[tid + t * 256] = v[t] * inv;
}

// ---------------------------------------------------------------------------
// Kernel 5: out = attn @ v. Wave owns a 16x64 output tile (4 accumulators);
// A = prob rows converted fp32->bf16 on the fly, B = v^T rows (contiguous).
// ---------------------------------------------------------------------------
__global__ __launch_bounds__(256)
void out_kernel(const float* __restrict__ attn, const __bf16* __restrict__ vt,
                float* __restrict__ out) {
    int bh = blockIdx.z;
    int b = bh >> 3, h = bh & 7;
    int wave = threadIdx.x >> 5;
    int lane = threadIdx.x & 31;
    int N = lane & 15, hf = lane >> 4;
    int i0 = (blockIdx.y * 8 + wave) * 16;

    const float*  ap = attn + (size_t)bh * L * L;
    const __bf16* vp = vt   + (size_t)bh * HD * L;

    v8f c0 = {}, c1 = {}, c2 = {}, c3 = {};
    for (int k0 = 0; k0 < L; k0 += 32) {
        v16bf a  = load_frag_f32(ap, L, i0, k0);
        v16bf b0 = load_frag_bf16(vp, L,  0, k0);
        v16bf b1 = load_frag_bf16(vp, L, 16, k0);
        v16bf b2 = load_frag_bf16(vp, L, 32, k0);
        v16bf b3 = load_frag_bf16(vp, L, 48, k0);
        c0 = wmma_bf16(a, b0, c0);
        c1 = wmma_bf16(a, b1, c1);
        c2 = wmma_bf16(a, b2, c2);
        c3 = wmma_bf16(a, b3, c3);
    }
#pragma unroll
    for (int e = 0; e < 8; ++e) {
        int i = i0 + e + (hf << 3);
        float* orow = out + ((size_t)b * L + i) * DM + h * HD;
        orow[ 0 + N] = c0[e];
        orow[16 + N] = c1[e];
        orow[32 + N] = c2[e];
        orow[48 + N] = c3[e];
    }
}

// ---------------------------------------------------------------------------
extern "C" void kernel_launch(void* const* d_in, const int* in_sizes, int n_in,
                              void* d_out, int out_size, void* d_ws, size_t ws_size,
                              hipStream_t stream) {
    const float* q    = (const float*)d_in[0];
    const float* k    = (const float*)d_in[1];
    const int*   mask = (const int*)  d_in[2];
    const float* Wq   = (const float*)d_in[3];
    const float* Wkv  = (const float*)d_in[4];
    const float* Wr   = (const float*)d_in[5];
    const float* rrb  = (const float*)d_in[6];   // r_r_bias
    const float* rwb  = (const float*)d_in[7];   // r_w_bias

    float* out  = (float*)d_out;                       // [B,L,D]
    float* attn = out + (size_t)NB * L * DM;           // [B,H,L,L]

    // Workspace layout (bf16): r[2L,64] | q1 | q2 | k | v^T  (~17.3 MB total)
    const size_t BHLHD = (size_t)NB * NH * L * HD;
    __bf16* rbf = (__bf16*)d_ws;
    __bf16* q1  = rbf + (size_t)2 * L * HD;
    __bf16* q2  = q1 + BHLHD;
    __bf16* kb  = q2 + BHLHD;
    __bf16* vt  = kb + BHLHD;

    build_r_kernel<<<2 * L, 64, 0, stream>>>(Wr, rbf);
    proj_kernel<<<dim3(12, L / 16, NB), 256, 0, stream>>>(q, k, Wq, Wkv, rrb, rwb,
                                                          q1, q2, kb, vt);
    scores_kernel<<<dim3(L / 128, L / 16, NB * NH), 256, 0, stream>>>(q1, q2, kb,
                                                                      rbf, attn);
    softmax_kernel<<<NB * NH * L, 256, 0, stream>>>(attn, mask);
    out_kernel<<<dim3(1, L / 16 / 8, NB * NH), 256, 0, stream>>>(attn, vt, out);
}
